// Attention_71811853189409
// MI455X (gfx1250) — compile-verified, matrix operand order checked by reading, not run on hardware
//
#include <hip/hip_runtime.h>
#include <hip/hip_bf16.h>

typedef __bf16 bf16_t;
typedef __attribute__((ext_vector_type(16))) __bf16 v16bf;
typedef __attribute__((ext_vector_type(8)))  __bf16 v8bf;
typedef __attribute__((ext_vector_type(8)))  float  v8f;
typedef __attribute__((ext_vector_type(4)))  unsigned uint32x4;
typedef __attribute__((ext_vector_type(8)))  int      int32x8;
typedef __attribute__((ext_vector_type(4)))  int      int32x4;

// ---- CDNA5 feature detection (compile-safe fallbacks) ----
#if defined(__has_builtin)
#  if __has_builtin(__builtin_amdgcn_tensor_load_to_lds)
#    define USE_TDM 1
#  endif
#  if __has_builtin(__builtin_amdgcn_global_load_async_to_lds_b128)
#    define USE_ASYNC 1
#  endif
#  if __has_builtin(__builtin_amdgcn_s_wait_asynccnt)
#    define HAS_WAIT_ASYNC_BUILTIN 1
#  endif
#endif
#ifndef USE_TDM
#  define USE_TDM 0
#endif
#ifndef USE_ASYNC
#  define USE_ASYNC 0
#endif

#define CAT16(lo, hi) __builtin_shufflevector((lo), (hi), 0,1,2,3,4,5,6,7,8,9,10,11,12,13,14,15)

static __device__ __forceinline__ v8f wmma_bf16(v16bf a, v16bf b, v8f c) {
  // D = A(16x32 bf16) * B(32x16 bf16) + C(16x16 f32)
  return __builtin_amdgcn_wmma_f32_16x16x32_bf16(false, a, false, b, (short)0, c, false, false);
}

#if USE_ASYNC
static __device__ __forceinline__ void wait_async0() {
#if defined(HAS_WAIT_ASYNC_BUILTIN)
  __builtin_amdgcn_s_wait_asynccnt(0);
#else
  asm volatile("s_wait_asynccnt 0" ::: "memory");
#endif
}
#endif

// ---------------------------------------------------------------------------
// fp32 -> bf16 conversion (grid-stride)
// ---------------------------------------------------------------------------
__global__ __launch_bounds__(256) void cvt_f32_bf16(const float* __restrict__ src,
                                                    bf16_t* __restrict__ dst,
                                                    size_t n) {
  size_t i = (size_t)blockIdx.x * blockDim.x + threadIdx.x;
  size_t stride = (size_t)gridDim.x * blockDim.x;
  for (; i < n; i += stride) dst[i] = (bf16_t)src[i];
}

// ---------------------------------------------------------------------------
// Generic bf16 GEMM: C[M,N] = A[M,K] * B[K,N]
// A row-major bf16, B row-major bf16. OUT_MODE 0: bf16 out. 1: f32 out + bias.
// Block: 256 threads (8 waves), tile 128x128, K-step 32.
// A-tile staged via TDM (tensor_load_to_lds) when available; B-tile transposed
// manually into LDS (TDM cannot transpose).
// ---------------------------------------------------------------------------
template <int OUT_MODE>
__global__ __launch_bounds__(256) void gemm_bf16(const bf16_t* __restrict__ A,
                                                 const bf16_t* __restrict__ Bm,
                                                 void* __restrict__ Cm,
                                                 const float* __restrict__ bias,
                                                 int M, int N, int K,
                                                 long long strideA_b,
                                                 long long strideB_b,
                                                 long long strideC_b) {
  __shared__ bf16_t As[128 * 40];  // [m][k], row stride 40 (80B, keeps 16B align)
  __shared__ bf16_t Bs[128 * 40];  // [n][k] (transposed), row stride 40

  const int b = blockIdx.z;
  A  += (size_t)b * strideA_b;
  Bm += (size_t)b * strideB_b;

  const int m0 = blockIdx.y * 128;
  const int n0 = blockIdx.x * 128;
  const int tid  = threadIdx.x;
  const int lane = tid & 31;
  const int wave = tid >> 5;
  const int waveM = wave >> 1;   // 0..3
  const int waveN = wave & 1;    // 0..1
  const int hf  = lane >> 4;     // lane half
  const int l16 = lane & 15;

  v8f acc[2][4];
#pragma unroll
  for (int mi = 0; mi < 2; ++mi)
#pragma unroll
    for (int ni = 0; ni < 4; ++ni)
      acc[mi][ni] = (v8f){0.f, 0.f, 0.f, 0.f, 0.f, 0.f, 0.f, 0.f};

  for (int k0 = 0; k0 < K; k0 += 32) {
    // ---- stage A tile: 128 rows x 32 cols (64B rows + 16B LDS pad) ----
#if USE_TDM
    if (tid == 0) {
      // LDS byte offset of As within the workgroup's LDS allocation
      unsigned lds_off = (unsigned)(unsigned long long)
          (__attribute__((address_space(3))) char*)As;
      unsigned long long ga =
          (unsigned long long)(A + (size_t)m0 * K + k0);
      // ---- D# group 0: count=1, lds_addr, 57b global_addr, type=2 ----
      uint32x4 g0;
      g0[0] = 1u;                                   // count=1, user mode
      g0[1] = lds_off;                              // lds_addr [63:32]
      g0[2] = (unsigned)ga;                         // global_addr [95:64]
      g0[3] = (unsigned)((ga >> 32) & 0x01FFFFFFu)  // global_addr [120:96]
              | (2u << 30);                         // type=2 ("image")
      // ---- D# group 1 ----
      // data_size=1 (2B), pad_enable, pad_interval=3 (16 DW), pad_amount=3 (4 DW)
      int32x8 g1;
      g1[0] = (int)((1u << 16) | (1u << 20) | (3u << 22) | (3u << 25));
      g1[1] = (int)(((unsigned)K & 0xFFFFu) << 16);          // tensor_dim0[15:0]
      g1[2] = (int)((((unsigned)K >> 16) & 0xFFFFu)          // tensor_dim0[31:16]
                    | (((unsigned)M & 0xFFFFu) << 16));      // tensor_dim1[15:0]
      g1[3] = (int)((((unsigned)M >> 16) & 0xFFFFu)          // tensor_dim1[31:16]
                    | (32u << 16));                          // tile_dim0 = 32
      g1[4] = (int)128u;                                     // tile_dim1=128, tile_dim2=0
      g1[5] = (int)(unsigned)K;                              // tensor_dim0_stride[31:0]
      g1[6] = 0;                                             // stride hi / dim1_stride lo
      g1[7] = 0;
      int32x4 gz = (int32x4){0, 0, 0, 0};                    // groups 2/3: <=2D tensor
#if __clang_major__ >= 23
      int32x8 gz8 = (int32x8){0, 0, 0, 0, 0, 0, 0, 0};
      __builtin_amdgcn_tensor_load_to_lds(g0, g1, gz, gz, gz8, 0);
#else
      __builtin_amdgcn_tensor_load_to_lds(g0, g1, gz, gz, 0);
#endif
      __builtin_amdgcn_s_wait_tensorcnt((short)0);
    }
#elif USE_ASYNC
#pragma unroll
    for (int c = tid; c < 512; c += 256) {
      int r = c >> 2, cg = c & 3;
      __builtin_amdgcn_global_load_async_to_lds_b128(
          (const __attribute__((address_space(1))) void*)
              (A + (size_t)(m0 + r) * K + k0 + cg * 8),
          (__attribute__((address_space(3))) void*)(As + r * 40 + cg * 8),
          0, 0);
    }
#else
#pragma unroll
    for (int c = tid; c < 512; c += 256) {
      int r = c >> 2, cg = c & 3;  // 4 chunks of 8 bf16 per row
      v8bf va = *(const v8bf*)(A + (size_t)(m0 + r) * K + k0 + cg * 8);
      *(v8bf*)(As + r * 40 + cg * 8) = va;
    }
#endif
    // ---- stage B tile transposed: 32 k-rows x 128 n-cols -> Bs[n][k] ----
#pragma unroll
    for (int c = tid; c < 512; c += 256) {
      int kk = c >> 4, ng = c & 15;  // 16 chunks of 8 bf16 along N
      v8bf vb = *(const v8bf*)(Bm + (size_t)(k0 + kk) * N + n0 + ng * 8);
#pragma unroll
      for (int e = 0; e < 8; ++e) Bs[(ng * 8 + e) * 40 + kk] = vb[e];
    }
    if (k0 + 32 < K) {
      __builtin_prefetch(Bm + (size_t)(k0 + 32 + (tid >> 4)) * N + n0, 0, 1);
    }
#if USE_TDM
    // tensor wait already executed by issuing wave before the barrier
#elif USE_ASYNC
    wait_async0();
#endif
    __syncthreads();

#pragma unroll
    for (int mi = 0; mi < 2; ++mi) {
      int arow = waveM * 32 + mi * 16 + l16;
      v8bf alo = *(const v8bf*)(As + arow * 40 + hf * 8);
      v8bf ahi = *(const v8bf*)(As + arow * 40 + 16 + hf * 8);
      v16bf afrag = CAT16(alo, ahi);
#pragma unroll
      for (int ni = 0; ni < 4; ++ni) {
        int bcol = waveN * 64 + ni * 16 + l16;
        v8bf blo = *(const v8bf*)(Bs + bcol * 40 + hf * 8);
        v8bf bhi = *(const v8bf*)(Bs + bcol * 40 + 16 + hf * 8);
        v16bf bfrag = CAT16(blo, bhi);
        acc[mi][ni] = wmma_bf16(afrag, bfrag, acc[mi][ni]);
      }
    }
    __syncthreads();
  }

  // ---- epilogue: v8f element r -> (M = hf*8 + r, N = l16) ----
#pragma unroll
  for (int mi = 0; mi < 2; ++mi) {
#pragma unroll
    for (int ni = 0; ni < 4; ++ni) {
      int rbase = m0 + waveM * 32 + mi * 16 + hf * 8;
      int col   = n0 + waveN * 64 + ni * 16 + l16;
#pragma unroll
      for (int r = 0; r < 8; ++r) {
        int row = rbase + r;
        float v = acc[mi][ni][r];
        if (OUT_MODE == 1) {
          ((float*)Cm)[(size_t)b * strideC_b + (size_t)row * N + col] = v + bias[row];
        } else {
          ((bf16_t*)Cm)[(size_t)b * strideC_b + (size_t)row * N + col] = (bf16_t)v;
        }
      }
    }
  }
}

// ---------------------------------------------------------------------------
// Per-row inverse L2 norm over n=4096 for q (rows 0..511) and k (rows 512..1023)
// One block per row (16*1024 rows).
// ---------------------------------------------------------------------------
__global__ __launch_bounds__(256) void rownorm_kernel(const bf16_t* __restrict__ qkv,
                                                      float* __restrict__ invnorm) {
  const int rowid = blockIdx.x;
  const int b = rowid >> 10;
  const int o = rowid & 1023;
  const bf16_t* p = qkv + (size_t)b * 1536 * 4096 + (size_t)o * 4096;
  float s = 0.f;
  for (int i = threadIdx.x; i < 4096; i += 256) {
    float v = (float)p[i];
    s += v * v;
  }
  __shared__ float red[256];
  red[threadIdx.x] = s;
  __syncthreads();
  for (int off = 128; off > 0; off >>= 1) {
    if (threadIdx.x < off) red[threadIdx.x] += red[threadIdx.x + off];
    __syncthreads();
  }
  if (threadIdx.x == 0) {
    float nrm = sqrtf(red[0]);
    invnorm[rowid] = 1.0f / fmaxf(nrm, 1e-12f);
  }
}

// ---------------------------------------------------------------------------
// Cross-covariance attention per (b,h): S = q*k^T (64x64, K=4096),
// scale by invq*invk*d^-0.5, softmax over cols, out = P * v (64x4096).
// 128 blocks, 256 threads (8 waves).
// ---------------------------------------------------------------------------
__global__ __launch_bounds__(256) void attn_kernel(const bf16_t* __restrict__ qkv,
                                                   const float* __restrict__ invnorm,
                                                   bf16_t* __restrict__ aout) {
  const int bh = blockIdx.x;
  const int b = bh >> 3, h = bh & 7;
  const size_t qkvb = (size_t)b * 1536 * 4096;
  const bf16_t* qp = qkv + qkvb + (size_t)(h * 64) * 4096;
  const bf16_t* kp = qkv + qkvb + (size_t)(512 + h * 64) * 4096;
  const bf16_t* vp = qkv + qkvb + (size_t)(1024 + h * 64) * 4096;
  const float* invq = invnorm + b * 1024 + h * 64;
  const float* invk = invnorm + b * 1024 + 512 + h * 64;

  __shared__ float  S[64 * 68];     // fp32 scores, row stride 68
  __shared__ bf16_t P[64 * 72];     // softmaxed probs, row stride 72 (144B -> 16B align)
  __shared__ bf16_t Vt[128 * 72];   // V transposed chunk: [n][k], row stride 72

  const int tid = threadIdx.x;
  const int lane = tid & 31;
  const int wave = tid >> 5;
  const int hf = lane >> 4;
  const int l16 = lane & 15;

  // ---------------- Phase 1: S = q * k^T over K = 4096 ----------------
  {
    const int waveM = wave >> 1;  // 0..3 -> 16-row strip
    const int waveN = wave & 1;   // 0..1 -> 32-col strip
    v8f acc[2] = {(v8f){0.f,0.f,0.f,0.f,0.f,0.f,0.f,0.f},
                  (v8f){0.f,0.f,0.f,0.f,0.f,0.f,0.f,0.f}};
    const int arow = waveM * 16 + l16;
    for (int k0 = 0; k0 < 4096; k0 += 32) {
      v8bf alo = *(const v8bf*)(qp + (size_t)arow * 4096 + k0 + hf * 8);
      v8bf ahi = *(const v8bf*)(qp + (size_t)arow * 4096 + k0 + 16 + hf * 8);
      v16bf afrag = CAT16(alo, ahi);
#pragma unroll
      for (int ni = 0; ni < 2; ++ni) {
        int brow = waveN * 32 + ni * 16 + l16;
        v8bf blo = *(const v8bf*)(kp + (size_t)brow * 4096 + k0 + hf * 8);
        v8bf bhi = *(const v8bf*)(kp + (size_t)brow * 4096 + k0 + 16 + hf * 8);
        v16bf bfrag = CAT16(blo, bhi);
        acc[ni] = wmma_bf16(afrag, bfrag, acc[ni]);
      }
    }
#pragma unroll
    for (int ni = 0; ni < 2; ++ni) {
#pragma unroll
      for (int r = 0; r < 8; ++r) {
        int row = waveM * 16 + hf * 8 + r;
        int col = waveN * 32 + ni * 16 + l16;
        S[row * 68 + col] = acc[ni][r] * invq[row] * invk[col] * 0.125f;
      }
    }
  }
  __syncthreads();

  // ---------------- softmax over 64 cols, one thread per row ----------------
  if (tid < 64) {
    float mx = -3.4e38f;
    for (int e = 0; e < 64; ++e) mx = fmaxf(mx, S[tid * 68 + e]);
    float sum = 0.f;
    for (int e = 0; e < 64; ++e) {
      float ex = __expf(S[tid * 68 + e] - mx);
      S[tid * 68 + e] = ex;
      sum += ex;
    }
    float inv = 1.0f / sum;
    for (int e = 0; e < 64; ++e) P[tid * 72 + e] = (bf16_t)(S[tid * 68 + e] * inv);
  }
  __syncthreads();

  // ---------------- Phase 2: out = P(64x64) * v(64x4096) ----------------
  const int waveM2 = wave & 3;   // 0..3 -> 16-row strip
  const int waveN2 = wave >> 2;  // 0..1 -> 64-col half of 128-col chunk
  bf16_t* outp = aout + (size_t)b * 512 * 4096 + (size_t)(h * 64) * 4096;

  for (int nc = 0; nc < 4096; nc += 128) {
    // stage V chunk transposed: v[64][nc..nc+128] -> Vt[n][k]
#pragma unroll
    for (int c = tid; c < 1024; c += 256) {
      int kk = c >> 4, ng = c & 15;
      v8bf vv = *(const v8bf*)(vp + (size_t)kk * 4096 + nc + ng * 8);
#pragma unroll
      for (int e = 0; e < 8; ++e) Vt[(ng * 8 + e) * 72 + kk] = vv[e];
    }
    if (nc + 128 < 4096)
      __builtin_prefetch(vp + (size_t)(tid >> 2) * 4096 + nc + 128, 0, 1);
    __syncthreads();

#pragma unroll
    for (int ni = 0; ni < 4; ++ni) {
      int ncol = waveN2 * 64 + ni * 16;
      v8f oacc = (v8f){0.f,0.f,0.f,0.f,0.f,0.f,0.f,0.f};
#pragma unroll
      for (int ks = 0; ks < 2; ++ks) {
        int arow = waveM2 * 16 + l16;
        v8bf alo = *(const v8bf*)(P + arow * 72 + ks * 32 + hf * 8);
        v8bf ahi = *(const v8bf*)(P + arow * 72 + ks * 32 + 16 + hf * 8);
        v16bf afrag = CAT16(alo, ahi);
        int bcol = ncol + l16;
        v8bf blo = *(const v8bf*)(Vt + bcol * 72 + ks * 32 + hf * 8);
        v8bf bhi = *(const v8bf*)(Vt + bcol * 72 + ks * 32 + 16 + hf * 8);
        v16bf bfrag = CAT16(blo, bhi);
        oacc = wmma_bf16(afrag, bfrag, oacc);
      }
#pragma unroll
      for (int r = 0; r < 8; ++r) {
        int row = waveM2 * 16 + hf * 8 + r;
        int col = nc + ncol + l16;
        outp[(size_t)row * 4096 + col] = (bf16_t)oacc[r];
      }
    }
    __syncthreads();
  }
}

// ---------------------------------------------------------------------------
// Host launcher
// ---------------------------------------------------------------------------
extern "C" void kernel_launch(void* const* d_in, const int* in_sizes, int n_in,
                              void* d_out, int out_size, void* d_ws, size_t ws_size,
                              hipStream_t stream) {
  const float* x      = (const float*)d_in[0];  // (16,512,64,64)
  const float* w_qkv  = (const float*)d_in[1];  // (1536,512)
  const float* w_proj = (const float*)d_in[2];  // (512,512)
  const float* b_proj = (const float*)d_in[3];  // (512,)
  float* out = (float*)d_out;

  const int Bn = 16, C = 512, Nn = 4096, M1 = 1536;

  char* ws = (char*)d_ws;
  size_t off = 0;
  bf16_t* xbf = (bf16_t*)(ws + off);   off += (size_t)Bn * C * Nn * 2;   // 64 MiB
  bf16_t* w1bf = (bf16_t*)(ws + off);  off += (size_t)M1 * C * 2;
  bf16_t* w2bf = (bf16_t*)(ws + off);  off += (size_t)C * C * 2;
  bf16_t* qkvbf = (bf16_t*)(ws + off); off += (size_t)Bn * M1 * Nn * 2;  // 192 MiB
  float*  invn = (float*)(ws + off);   off += (size_t)Bn * 1024 * 4;
  bf16_t* aoutbf = (bf16_t*)(ws + off); off += (size_t)Bn * C * Nn * 2;  // 64 MiB

  // 1) convert inputs to bf16
  cvt_f32_bf16<<<4096, 256, 0, stream>>>(x, xbf, (size_t)Bn * C * Nn);
  cvt_f32_bf16<<<256, 256, 0, stream>>>(w_qkv, w1bf, (size_t)M1 * C);
  cvt_f32_bf16<<<128, 256, 0, stream>>>(w_proj, w2bf, (size_t)C * C);

  // 2) qkv = w_qkv * x   (per batch: [1536x512]x[512x4096])
  gemm_bf16<0><<<dim3(Nn / 128, M1 / 128, Bn), 256, 0, stream>>>(
      w1bf, xbf, qkvbf, nullptr, M1, Nn, C,
      0LL, (long long)C * Nn, (long long)M1 * Nn);

  // 3) inverse L2 norms for q,k rows
  rownorm_kernel<<<Bn * 1024, 256, 0, stream>>>(qkvbf, invn);

  // 4) attention per (b,h)
  attn_kernel<<<Bn * 8, 256, 0, stream>>>(qkvbf, invn, aoutbf);

  // 5) out = w_proj * attn_out + bias  (fp32 output)
  gemm_bf16<1><<<dim3(Nn / 128, C / 128, Bn), 256, 0, stream>>>(
      w2bf, aoutbf, out, b_proj, C, Nn, C,
      0LL, (long long)C * Nn, (long long)C * Nn);
}
